// _Fp8EinsumWrapper_51934744543257
// MI455X (gfx1250) — 8-bit matrix operand order repaired and checked by reading, compiled, not run on hardware
//
#include <hip/hip_runtime.h>
#include <hip/hip_bf16.h>

typedef __attribute__((ext_vector_type(16))) int   v16i;
typedef __attribute__((ext_vector_type(8)))  float v8f;

static constexpr int M_ = 16384;
static constexpr int K_ = 4096;
static constexpr int N_ = 1024;
static constexpr int KS_ = K_ / 128;
#define E4M3_MAX 448.0f

#if defined(__has_builtin)
#  if __has_builtin(__builtin_amdgcn_cvt_pk_fp8_f32)
#    define USE_HW_CVT_FP8 1
#  endif
#endif

__device__ __forceinline__ unsigned int sw_e4m3(float x) {
    unsigned int u   = __float_as_uint(x);
    unsigned int sgn = (u >> 31) << 7;
    float ax = fabsf(x);
    if (ax < 0.0009765625f) return sgn;
    if (ax < 0.015625f) {
        unsigned int r = (unsigned int)rintf(ax * 512.0f);
        return sgn | r;
    }
    unsigned int b   = __float_as_uint(ax);
    unsigned int lsb = (b >> 20) & 1u;
    b += 0x7FFFFu + lsb;
    int e4 = (int)(b >> 23) - 127 + 7;
    unsigned int mant = (b >> 20) & 7u;
    if (e4 >= 16) return sgn | 0x7Eu;
    return sgn | ((unsigned int)e4 << 3) | mant;
}

__device__ __forceinline__ unsigned int pack4_e4m3(float x0, float x1, float x2, float x3) {
#ifdef USE_HW_CVT_FP8
    int p = 0;
    p = __builtin_amdgcn_cvt_pk_fp8_f32(x0, x1, p, false);
    p = __builtin_amdgcn_cvt_pk_fp8_f32(x2, x3, p, true);
    return (unsigned int)p;
#else
    return sw_e4m3(x0) | (sw_e4m3(x1) << 8) | (sw_e4m3(x2) << 16) | (sw_e4m3(x3) << 24);
#endif
}

__device__ __forceinline__ float clip448(float x) {
    return fminf(fmaxf(x, -E4M3_MAX), E4M3_MAX);
}


__global__ __launch_bounds__(256) void quant_a(const unsigned short* __restrict__ in,
                                               unsigned char* __restrict__ asw,
                                               const float* __restrict__ sptr) {
    const unsigned int tid = blockIdx.x * 256 + threadIdx.x;
    const int lane = tid & 31;
    const int c    = (tid >> 5) & 3;
    const int ks   = (tid >> 7) & (KS_ - 1);
    const int mt   = tid >> 12;
    const int lm   = lane & 15;
    const int g    = lane >> 4;

    const float inv = 1.0f / sptr[0];
    const size_t row = (size_t)(mt * 16 + lm) * K_;
    const int kb = ks * 128 + c * 32 + g * 16;

    uint4 r0 = *(const uint4*)(in + row + kb);
    uint4 r1 = *(const uint4*)(in + row + kb + 8);
    unsigned int w[8] = {r0.x, r0.y, r0.z, r0.w, r1.x, r1.y, r1.z, r1.w};
    float f[16];
#pragma unroll
    for (int j = 0; j < 8; j++) {
        f[2 * j]     = clip448(__uint_as_float(w[j] << 16) * inv);
        f[2 * j + 1] = clip448(__uint_as_float(w[j] & 0xFFFF0000u) * inv);
    }
    uint4 o;
    o.x = pack4_e4m3(f[0],  f[1],  f[2],  f[3]);
    o.y = pack4_e4m3(f[4],  f[5],  f[6],  f[7]);
    o.z = pack4_e4m3(f[8],  f[9],  f[10], f[11]);
    o.w = pack4_e4m3(f[12], f[13], f[14], f[15]);
    *(uint4*)(asw + (size_t)tid * 16) = o;
}

__global__ __launch_bounds__(256) void quant_bt(const float* __restrict__ rhs,
                                                unsigned char* __restrict__ bsw,
                                                const float* __restrict__ sptr) {
    __shared__ float tile[128][33];
    const int n0 = blockIdx.x * 32;
    const int ks = blockIdx.y;
    const float inv = 1.0f / sptr[0];
#pragma unroll
    for (int r = 0; r < 16; r++) {
        const int kk = threadIdx.y + 8 * r;
        tile[kk][threadIdx.x] =
            rhs[(size_t)(ks * 128 + kk) * N_ + n0 + threadIdx.x];
    }
    __syncthreads();
    const int t    = threadIdx.y * 32 + threadIdx.x;
    const int lane = t & 31;
    const int q    = (t >> 5) & 3;
    const int tn   = t >> 7;
    const int lm   = lane & 15;
    const int g    = lane >> 4;
    const int col  = tn * 16 + lm;
    const int kr   = q * 32 + g * 16;
    float f[16];
#pragma unroll
    for (int i = 0; i < 16; i++) f[i] = clip448(tile[kr + i][col] * inv);
    uint4 o;
    o.x = pack4_e4m3(f[0],  f[1],  f[2],  f[3]);
    o.y = pack4_e4m3(f[4],  f[5],  f[6],  f[7]);
    o.z = pack4_e4m3(f[8],  f[9],  f[10], f[11]);
    o.w = pack4_e4m3(f[12], f[13], f[14], f[15]);
    const size_t nt = (size_t)(blockIdx.x * 2 + tn);
    *(uint4*)(bsw + (((nt * KS_ + ks) * 4 + q) * 32 + lane) * 16) = o;
}

__global__ __launch_bounds__(256) void fp8_gemm_wmma(
        const unsigned char* __restrict__ Asw,
        const unsigned char* __restrict__ Bsw,
        const float* __restrict__ sA, const float* __restrict__ sB,
        __hip_bfloat16* __restrict__ out) {
    const int lane = threadIdx.x & 31;
    const int wave = threadIdx.x >> 5;
    const int wm = wave & 3;
    const int wn = wave >> 2;
    const int mt0 = blockIdx.y * 8 + wm * 2;
    const int nt0 = blockIdx.x * 8 + wn * 4;
    const int g  = lane >> 4;
    const int lm = lane & 15;

    const unsigned char* ap0 = Asw + (size_t)(mt0    ) * KS_ * 2048 + lane * 16;
    const unsigned char* ap1 = Asw + (size_t)(mt0 + 1) * KS_ * 2048 + lane * 16;
    const unsigned char* bp[4];
#pragma unroll
    for (int ni = 0; ni < 4; ni++)
        bp[ni] = Bsw + (size_t)(nt0 + ni) * KS_ * 2048 + lane * 16;

    v8f acc[2][4];
#pragma unroll
    for (int i = 0; i < 2; i++)
#pragma unroll
        for (int j = 0; j < 4; j++) acc[i][j] = {};

    for (int ks = 0; ks < KS_; ks++) {
        const size_t off = (size_t)ks * 2048;
        v16i af[2], bf[4];
#pragma unroll
        for (int c = 0; c < 4; c++) {
            uint4 v0 = *(const uint4*)(ap0 + off + c * 512);
            uint4 v1 = *(const uint4*)(ap1 + off + c * 512);
            af[0][4*c] = (int)v0.x; af[0][4*c+1] = (int)v0.y;
            af[0][4*c+2] = (int)v0.z; af[0][4*c+3] = (int)v0.w;
            af[1][4*c] = (int)v1.x; af[1][4*c+1] = (int)v1.y;
            af[1][4*c+2] = (int)v1.z; af[1][4*c+3] = (int)v1.w;
        }
#pragma unroll
        for (int ni = 0; ni < 4; ni++) {
#pragma unroll
            for (int q = 0; q < 4; q++) {
                uint4 v = *(const uint4*)(bp[ni] + off + q * 512);
                bf[ni][4*q] = (int)v.x; bf[ni][4*q+1] = (int)v.y;
                bf[ni][4*q+2] = (int)v.z; bf[ni][4*q+3] = (int)v.w;
            }
        }
#pragma unroll
        for (int mi = 0; mi < 2; mi++)
#pragma unroll
            for (int ni = 0; ni < 4; ni++)
                acc[mi][ni] = __builtin_amdgcn_wmma_f32_16x16x128_fp8_fp8(
                    af[mi], bf[ni], (short)0, acc[mi][ni], false, false);
    }

    const float s = sA[0] * sB[0];
#pragma unroll
    for (int mi = 0; mi < 2; mi++) {
#pragma unroll
        for (int ni = 0; ni < 4; ni++) {
            const int col = (nt0 + ni) * 16 + lm;
#pragma unroll
            for (int r = 0; r < 8; r++) {
                const int row = (mt0 + mi) * 16 + g * 8 + r;
                out[(size_t)row * N_ + col] = __float2bfloat16(acc[mi][ni][r] * s);
            }
        }
    }
}

extern "C" void kernel_launch(void* const* d_in, const int* in_sizes, int n_in,
                              void* d_out, int out_size, void* d_ws, size_t ws_size,
                              hipStream_t stream) {
    const unsigned short* lhs  = (const unsigned short*)d_in[0];
    const float*          rhs  = (const float*)d_in[1];
    const float*          s_in = (const float*)d_in[2];
    const float*          s_kr = (const float*)d_in[3];
    __hip_bfloat16*       out  = (__hip_bfloat16*)d_out;

    unsigned char* a_q = (unsigned char*)d_ws;
    unsigned char* b_q = a_q + (size_t)M_ * K_;

    quant_a<<<(M_ / 16) * KS_ * 4 * 32 / 256, 256, 0, stream>>>(lhs, a_q, s_in);
    quant_bt<<<dim3(N_ / 32, KS_), dim3(32, 8), 0, stream>>>(rhs, b_q, s_kr);
    fp8_gemm_wmma<<<dim3(N_ / 128, M_ / 128), 256, 0, stream>>>(a_q, b_q, s_in, s_kr, out);
}
